// ConvLayer_64115271795212
// MI455X (gfx1250) — matrix operand order repaired and checked by reading, not compiled, not run on hardware
//
#include <hip/hip_runtime.h>
#include <cmath>

typedef __attribute__((ext_vector_type(16))) _Float16 v16h;
typedef __attribute__((ext_vector_type(8)))  _Float16 v8h;
typedef __attribute__((ext_vector_type(4)))  _Float16 v4h;
typedef __attribute__((ext_vector_type(8)))  float    v8f;

#define N_ATOMS       100000
#define A_FEA         64
#define NBR_FEA       48
#define M_NBR         24
#define IN1           176          // 2A + F
#define OUT1          128          // 2A
#define KPAD          192          // 176 padded to 6 K-tiles of 32
#define ATOMS_PER_BLK 8
#define ROWS          (ATOMS_PER_BLK * M_NBR)   // 192
#define ROW_TILES    (ROWS / 16)                // 12
#define K_TILES       (KPAD / 32)               // 6
#define BN_EPS        1e-5f

#define USE_ASYNC 1

// Branchless, overflow-safe softplus: max(x,0) + log(1 + exp(-|x|)).
// Lowers to v_exp_f32 / v_log_f32 straight-line code (no EXEC-mask branches).
__device__ __forceinline__ float softplus_f(float x) {
    return fmaxf(x, 0.f) + __logf(1.f + __expf(-fabsf(x)));
}

// ---------------------------------------------------------------------------
// Prep: one-time f16 conversion of atom features and W1 (zero-padded K->192).
// ---------------------------------------------------------------------------
__global__ __launch_bounds__(256)
void prep_f16(const float* __restrict__ atom_in,
              const float* __restrict__ W1,
              _Float16* __restrict__ Ah,
              _Float16* __restrict__ W1h)
{
    const int total_a = N_ATOMS * A_FEA;
    int t = blockIdx.x * 256 + threadIdx.x;
    if (t < total_a) {
        Ah[t] = (_Float16)atom_in[t];
    } else if (t < total_a + OUT1 * KPAD) {
        int i = t - total_a;
        int o = i / KPAD, k = i - o * KPAD;
        W1h[i] = (k < IN1) ? (_Float16)W1[o * IN1 + k] : (_Float16)0.f;
    }
}

// ---------------------------------------------------------------------------
// Main fused kernel.
// ---------------------------------------------------------------------------
__global__ __launch_bounds__(256)
void cgcnn_conv_main(const _Float16* __restrict__ Ah,     // [N][64] f16
                     const float* __restrict__ nbr_fea,
                     const int*   __restrict__ nbr_idx,
                     const _Float16* __restrict__ W1h,    // [128][192] f16
                     const float* __restrict__ b1,
                     const float* __restrict__ W2,
                     const float* __restrict__ b2,
                     float* __restrict__ h_out,
                     float* __restrict__ stat_sum,
                     float* __restrict__ stat_sq)
{
    extern __shared__ char smem[];
    _Float16* Xs   = (_Float16*)smem;                     // ROWS * KPAD f16
    _Float16* Ws   = Xs + ROWS * KPAD;                    // OUT1 * KPAD f16
    float*    sums = (float*)(Ws + OUT1 * KPAD);          // [8][128]
    float*    hsh  = sums + ATOMS_PER_BLK * OUT1;         // [8][64]

    const int tid   = threadIdx.x;
    const int atom0 = blockIdx.x * ATOMS_PER_BLK;

    {   // zero per-atom accumulators: 8*128 floats == 256 float4 stores
        float4 z4 = make_float4(0.f, 0.f, 0.f, 0.f);
        ((float4*)sums)[tid] = z4;
    }

    if (tid < ROWS) {
        // Waves 0..5: one gathered X row per thread, fully vectorized.
        const int r = tid;
        const int a = r / M_NBR;
        const int m = r - a * M_NBR;
        const int n = atom0 + a;
        _Float16* dst = Xs + r * KPAD;

        const v8h* usrc = (const v8h*)(Ah + (size_t)n * A_FEA);
        v8h* d0 = (v8h*)dst;
#pragma unroll
        for (int i = 0; i < 8; ++i) d0[i] = usrc[i];

        const int nb = nbr_idx[n * M_NBR + m];
        const v8h* nsrc = (const v8h*)(Ah + (size_t)nb * A_FEA);
#pragma unroll
        for (int i = 0; i < 8; ++i) d0[8 + i] = nsrc[i];

        const float4* fsrc =
            (const float4*)(nbr_fea + ((size_t)n * M_NBR + m) * NBR_FEA);
        v4h* d1 = (v4h*)(dst + 2 * A_FEA);
#pragma unroll
        for (int i = 0; i < 12; ++i) {
            float4 f = fsrc[i];
            v4h h4;
            h4[0] = (_Float16)f.x; h4[1] = (_Float16)f.y;
            h4[2] = (_Float16)f.z; h4[3] = (_Float16)f.w;
            d1[i] = h4;
        }

        v8h z = {};
        ((v8h*)(dst + IN1))[0] = z;
        ((v8h*)(dst + IN1 + 8))[0] = z;
    } else {
        // Waves 6..7: stage 48KB of pre-converted W1h into LDS.
        const int j = tid - ROWS;                         // 0..63
#if USE_ASYNC
        const uint32_t lds_base = (uint32_t)(uintptr_t)Ws;
        for (int it = 0; it < 48; ++it) {
            uint32_t byte = (uint32_t)(it * 64 + j) * 16u;
            asm volatile("global_load_async_to_lds_b128 %0, %1, %2"
                         :: "v"(lds_base + byte), "v"(byte), "s"(W1h)
                         : "memory");
        }
        asm volatile("s_wait_asynccnt 0" ::: "memory");
#else
        for (int it = 0; it < 48; ++it) {
            uint32_t byte = (uint32_t)(it * 64 + j) * 16u;
            *(v8h*)((char*)Ws + byte) = *(const v8h*)((const char*)W1h + byte);
        }
#endif
    }
    __syncthreads();

    const int wave = tid >> 5;
    const int lane = tid & 31;
    const int ocol = lane & 15;
    const int o    = wave * 16 + ocol;
    const float bias = b1[o];
    const int  aoff = (lane < 16) ? 0 : 8;

    // B fragments (this wave's 16-column slice), resident across row tiles.
    v16h bf[K_TILES];
#pragma unroll
    for (int kt = 0; kt < K_TILES; ++kt) {
        int kb = kt * 32 + ((lane < 16) ? 0 : 8);
        const v8h* p = (const v8h*)(Ws + o * KPAD + kb);
        v8h lo = p[0], hi = p[2];
        v16h b;
#pragma unroll
        for (int e = 0; e < 8; ++e) { b[e] = lo[e]; b[8 + e] = hi[e]; }
        bf[kt] = b;
    }

#pragma unroll
    for (int rt = 0; rt < ROW_TILES; ++rt) {
        v8f acc = {};
        const int row = rt * 16 + ocol;
#pragma unroll
        for (int kt = 0; kt < K_TILES; ++kt) {
            const _Float16* ap = Xs + row * KPAD + kt * 32 + aoff;
            v8h lo = *(const v8h*)ap;
            v8h hi = *(const v8h*)(ap + 16);
            v16h af;
#pragma unroll
            for (int e = 0; e < 8; ++e) { af[e] = lo[e]; af[8 + e] = hi[e]; }
            acc = __builtin_amdgcn_wmma_f32_16x16x32_f16(
                      false, af, false, bf[kt], (short)0, acc, false, false);
        }

        // Epilogue: this lane's 8 consecutive rows (base..base+7) span at most
        // two atoms. One division per tile, two ds_add_f32 instead of eight.
        const int base = rt * 16 + aoff;       // first row of this lane group
        const int a0   = base / M_NBR;
        const int m0   = base - a0 * M_NBR;
        const int rem  = M_NBR - m0;           // rows left in atom a0 (>=1)
        float p0 = 0.f, p1 = 0.f;
#pragma unroll
        for (int v = 0; v < 8; ++v) {
            float g    = acc[v] + bias;
            float relu = fmaxf(g, 0.f);
            float sp   = relu + __logf(1.f + __expf(-fabsf(g)));
            int   m    = (v < rem) ? (m0 + v) : (v - rem);
            float act  = (m < (M_NBR / 2)) ? relu : sp;
            p0 += (v < rem) ? act : 0.f;
            p1 += (v < rem) ? 0.f : act;
        }
        atomicAdd(&sums[a0 * OUT1 + o], p0);
        const int a1 = (a0 < ATOMS_PER_BLK - 1) ? a0 + 1 : a0;  // p1==0 when clamped
        atomicAdd(&sums[a1 * OUT1 + o], p1);
    }
    __syncthreads();

    // GEMM2 (fp32): h[a][j] = b2[j] + sums[a][:] . W2[j][:]
#pragma unroll
    for (int p = 0; p < 2; ++p) {
        int idx = tid + p * 256;
        int a = idx >> 6, j = idx & 63;
        const float4* srow = (const float4*)(sums + a * OUT1);
        const float4* wrow = (const float4*)(W2 + j * OUT1);
        float acc2 = b2[j];
#pragma unroll
        for (int q = 0; q < OUT1 / 4; ++q) {
            float4 s4 = srow[q], w4 = wrow[q];
            acc2 += s4.x * w4.x + s4.y * w4.y + s4.z * w4.z + s4.w * w4.w;
        }
        h_out[(size_t)(atom0 + a) * A_FEA + j] = acc2;
        hsh[idx] = acc2;
    }
    __syncthreads();

    if (tid < A_FEA) {
        float s = 0.f, sq = 0.f;
#pragma unroll
        for (int a = 0; a < ATOMS_PER_BLK; ++a) {
            float hv = hsh[a * A_FEA + tid];
            s  += hv;
            sq += hv * hv;
        }
        atomicAdd(&stat_sum[tid], s);
        atomicAdd(&stat_sq[tid],  sq);
    }
}

__global__ void zero_stats(float* s) {
    if (threadIdx.x < 128) s[threadIdx.x] = 0.f;
}

__global__ void finalize_stats(const float* __restrict__ ssum,
                               const float* __restrict__ ssq,
                               float* __restrict__ mean,
                               float* __restrict__ inv)
{
    int j = threadIdx.x;
    if (j < A_FEA) {
        float m = ssum[j] * (1.f / (float)N_ATOMS);
        float v = ssq[j]  * (1.f / (float)N_ATOMS) - m * m;
        mean[j] = m;
        inv[j]  = rsqrtf(v + BN_EPS);
    }
}

__global__ __launch_bounds__(256)
void bn_softplus(const float* __restrict__ h,
                 const float* __restrict__ atom_in,
                 const float* __restrict__ gamma,
                 const float* __restrict__ beta,
                 const float* __restrict__ mean,
                 const float* __restrict__ inv,
                 float* __restrict__ out)
{
    int tid = blockIdx.x * 256 + threadIdx.x;
    if (tid >= N_ATOMS * A_FEA) return;
    int j = tid & (A_FEA - 1);
    float y = gamma[j] * (h[tid] - mean[j]) * inv[j] + beta[j] + atom_in[tid];
    out[tid] = softplus_f(y);
}

extern "C" void kernel_launch(void* const* d_in, const int* in_sizes, int n_in,
                              void* d_out, int out_size, void* d_ws, size_t ws_size,
                              hipStream_t stream)
{
    const float* atom_in = (const float*)d_in[0];
    const float* nbr_fea = (const float*)d_in[1];
    const int*   nbr_idx = (const int*)d_in[2];
    const float* W1      = (const float*)d_in[3];
    const float* b1      = (const float*)d_in[4];
    const float* W2      = (const float*)d_in[5];
    const float* b2      = (const float*)d_in[6];
    const float* gamma   = (const float*)d_in[7];
    const float* beta    = (const float*)d_in[8];
    float* out = (float*)d_out;

    float* ws    = (float*)d_ws;
    float* h     = ws;                                  // N*64 f32
    float* ssum  = ws + (size_t)N_ATOMS * A_FEA;        // 64
    float* ssq   = ssum + 64;                           // 64
    float* smean = ssq  + 64;                           // 64
    float* sinv  = smean + 64;                          // 64
    _Float16* Ah  = (_Float16*)(sinv + 64);             // N*64 f16
    _Float16* W1h = Ah + (size_t)N_ATOMS * A_FEA;       // 128*192 f16

    zero_stats<<<1, 128, 0, stream>>>(ssum);

    const int prep_total = N_ATOMS * A_FEA + OUT1 * KPAD;
    prep_f16<<<(prep_total + 255) / 256, 256, 0, stream>>>(atom_in, W1, Ah, W1h);

    size_t lds_bytes = (size_t)(ROWS * KPAD + OUT1 * KPAD) * sizeof(_Float16)
                     + (size_t)(ATOMS_PER_BLK * OUT1 + ATOMS_PER_BLK * A_FEA) * sizeof(float);

    cgcnn_conv_main<<<N_ATOMS / ATOMS_PER_BLK, 256, lds_bytes, stream>>>(
        Ah, nbr_fea, nbr_idx, W1h, b1, W2, b2, h, ssum, ssq);

    finalize_stats<<<1, 64, 0, stream>>>(ssum, ssq, smean, sinv);

    bn_softplus<<<(N_ATOMS * A_FEA + 255) / 256, 256, 0, stream>>>(
        h, atom_in, gamma, beta, smean, sinv, out);
}
